// MultiHeadSelfAttention_60868276519484
// MI455X (gfx1250) — compile-verified, matrix operand order checked by reading, not run on hardware
//
#include <hip/hip_runtime.h>
#include <hip/hip_bf16.h>

// ---------------- problem constants ----------------
#define D_MODEL   768
#define NUM_HEADS 12
#define HEAD_DIM  64
#define SEQ       2048
#define BATCH     2
#define QKV_N     (3 * D_MODEL)     // 2304
#define MROWS     (BATCH * SEQ)     // 4096

typedef __attribute__((ext_vector_type(16))) __bf16 v16bf;
typedef __attribute__((ext_vector_type(8)))  float  v8f;

// ---------------- helpers ----------------
__device__ __forceinline__ unsigned short f2bf(float f) {
  unsigned int u = __float_as_uint(f);
  u += 0x7fffu + ((u >> 16) & 1u);           // round-to-nearest-even
  return (unsigned short)(u >> 16);
}

// Load one 16x32 bf16 WMMA operand fragment from a row-major bf16 matrix.
// A-operand layout: lane = row (M), regs = K.  B-operand layout is identical
// with lane = column (N), regs = K — so for X @ W^T both operands are plain
// row-major fragment loads.
//   lanes 0-15 : K = col..col+7 (regs0-3), col+16..col+23 (regs4-7)
//   lanes 16-31: same rows, K shifted by +8 (caller adds kShift to col)
__device__ __forceinline__ v16bf load_frag(const unsigned short* p, int ld, int row, int col) {
  union { uint4 u[2]; v16bf v; } f;
  const uint4* q = reinterpret_cast<const uint4*>(p + (long)row * ld + col);
  f.u[0] = q[0];      // K = col .. col+7
  f.u[1] = q[2];      // K = col+16 .. col+23
  return f.v;
}

__device__ __forceinline__ v8f wmma_bf16(v16bf a, v16bf b, v8f c) {
  return __builtin_amdgcn_wmma_f32_16x16x32_bf16(false, a, false, b, (short)0, c, false, false);
}

// CDNA5 async global->LDS copy (16B per lane), tracked by ASYNCcnt.
// VDST operand carries the LDS byte address; low 32 bits of a generic LDS
// pointer are the raw LDS offset (aperture tag is in the high dword).
__device__ __forceinline__ void async_copy16(void* lds_dst, const void* gsrc) {
  unsigned lds = (unsigned)(unsigned long long)lds_dst;
  unsigned long long ga = (unsigned long long)gsrc;
  asm volatile("global_load_async_to_lds_b128 %0, %1, off"
               :: "v"(lds), "v"(ga) : "memory");
}
__device__ __forceinline__ void wait_async0() {
  asm volatile("s_wait_asynccnt 0x0" ::: "memory");
}

// ---------------- kernel 0: f32 -> bf16 pack (4 elems/thread) ----------------
__global__ void __launch_bounds__(256)
cvt_bf16_kernel(const float* __restrict__ src, unsigned short* __restrict__ dst, int n4) {
  int i = blockIdx.x * 256 + threadIdx.x;
  if (i >= n4) return;
  float4 f = reinterpret_cast<const float4*>(src)[i];
  uint2 o;
  o.x = (unsigned)f2bf(f.x) | ((unsigned)f2bf(f.y) << 16);
  o.y = (unsigned)f2bf(f.z) | ((unsigned)f2bf(f.w) << 16);
  reinterpret_cast<uint2*>(dst)[i] = o;
}

// ---------------- kernel 1: QKV GEMM + fused RoPE + scatter ----------------
// qkv[m, n] = sum_k x[m,k] * Wqkv[n,k];  RoPE applied to q/k sections,
// results scattered bf16 into Q[b,h,s,d], K[b,h,s,d], Vt[b,h,d,s].
__global__ void __launch_bounds__(128)
qkv_rope_kernel(const unsigned short* __restrict__ xbf,
                const unsigned short* __restrict__ wbf,
                const int*            __restrict__ tpos,
                unsigned short* __restrict__ Qb,
                unsigned short* __restrict__ Kb,
                unsigned short* __restrict__ Vt) {
  const int lane   = threadIdx.x & 31;
  const int wave   = threadIdx.x >> 5;
  const int r      = lane & 15;
  const int hi     = lane >> 4;          // 0 | 1
  const int kShift = hi * 8;
  const int mBase  = blockIdx.x * 16;
  const int nBase  = blockIdx.y * 256 + wave * 64;

  v8f zero = {0.f,0.f,0.f,0.f,0.f,0.f,0.f,0.f};
  v8f acc[4] = {zero, zero, zero, zero};

  for (int kk = 0; kk < D_MODEL; kk += 32) {
    __builtin_prefetch(xbf + (long)(mBase + r) * D_MODEL + kk + 64, 0, 1);
    v16bf a = load_frag(xbf, D_MODEL, mBase + r, kk + kShift);
#pragma unroll
    for (int i = 0; i < 4; ++i) {
      v16bf b = load_frag(wbf, D_MODEL, nBase + i * 16 + r, kk + kShift);
      acc[i] = wmma_bf16(a, b, acc[i]);
    }
  }

#pragma unroll
  for (int i = 0; i < 4; ++i) {
    const int n0     = nBase + i * 16;
    const int which  = n0 / D_MODEL;              // 0=q 1=k 2=v (tile-uniform)
    const int within = n0 % D_MODEL;
    const int h      = within / HEAD_DIM;
    const int d      = (within % HEAD_DIM) + r;   // lane's feature index
#pragma unroll
    for (int j = 0; j < 8; ++j) {
      const int m = mBase + j + hi * 8;
      const int b = m / SEQ;
      const int s = m - b * SEQ;
      float v = acc[i][j];
      if (which < 2) {                            // RoPE on q and k
        const int pos = tpos[b * SEQ + s];
        // freq = 10000^(-(d&~1)/64) = exp(-(d&~1) * ln(1e4)/64)
        const float freq = __expf(-(float)(d & ~1) * 0.14391156516f);
        float sn, cs;
        __sincosf((float)pos * freq, &sn, &cs);
        const float partner = __shfl_xor(v, 1, 32);   // even<->odd feature
        v = v * cs + partner * ((d & 1) ? sn : -sn);
      }
      const unsigned short bv = f2bf(v);
      const long bh = (long)b * NUM_HEADS + h;
      if      (which == 0) Qb[(bh * SEQ + s) * HEAD_DIM + d] = bv;
      else if (which == 1) Kb[(bh * SEQ + s) * HEAD_DIM + d] = bv;
      else                 Vt[(bh * HEAD_DIM + d) * SEQ + s] = bv;   // transposed
    }
  }
}

// ---------------- kernel 2: causal flash attention ----------------
// Block = 4 waves = 64 query rows. Shared K/V tiles (32 keys) are staged in
// LDS once per block via async global->LDS copies (ASYNCcnt) and reused by
// all 4 waves: 4x less global K/V traffic, ~64 FLOP/B vs WMMA throughput.
__global__ void __launch_bounds__(128)
attn_kernel(const unsigned short* __restrict__ Qb,
            const unsigned short* __restrict__ Kb,
            const unsigned short* __restrict__ Vt,
            unsigned short* __restrict__ ctx) {
  __shared__ __align__(16) unsigned short kt[32][HEAD_DIM];   // keys x d   (4KB)
  __shared__ __align__(16) unsigned short vt[HEAD_DIM][32];   // d x keys   (4KB)
  __shared__ __align__(16) unsigned short pbuf[4][16][32];    // per-wave P (4KB)

  const int lane   = threadIdx.x & 31;
  const int wave   = threadIdx.x >> 5;
  const int r      = lane & 15;
  const int hi     = lane >> 4;
  const int kShift = hi * 8;

  const int bh  = blockIdx.y;
  const int b   = bh / NUM_HEADS;
  const int h   = bh % NUM_HEADS;
  const int q0B = blockIdx.x * 64;            // block's query base
  const int q0  = q0B + wave * 16;            // wave's query base

  const unsigned short* Qh = Qb + (long)bh * SEQ * HEAD_DIM;
  const unsigned short* Kh = Kb + (long)bh * SEQ * HEAD_DIM;
  const unsigned short* Vh = Vt + (long)bh * HEAD_DIM * SEQ;

  v16bf qf[2];
  qf[0] = load_frag(Qh, HEAD_DIM, q0 + r, 0  + kShift);
  qf[1] = load_frag(Qh, HEAD_DIM, q0 + r, 32 + kShift);

  v8f zero = {0.f,0.f,0.f,0.f,0.f,0.f,0.f,0.f};
  v8f o[4] = {zero, zero, zero, zero};
  float mrow[8], lrow[8];
#pragma unroll
  for (int j = 0; j < 8; ++j) { mrow[j] = -1e30f; lrow[j] = 0.f; }

  const float scale = 0.125f;                 // 1/sqrt(64)
  const int kend      = q0 + 16;              // this wave's causal bound
  const int blockKend = q0B + 64;             // block's causal bound

  for (int kb = 0; kb < blockKend; kb += 32) {
    __syncthreads();                          // previous tiles fully consumed
    // ---- stage K(32x64) and V^T(64x32) tiles: 2 async 16B chunks/thread ----
#pragma unroll
    for (int i = 0; i < 2; ++i) {
      const int c    = threadIdx.x + i * 128;        // 0..255
      const int krow = c >> 3, kcol = (c & 7) * 8;   // 8 chunks per K row
      async_copy16(&kt[krow][kcol],
                   Kh + (long)(kb + krow) * HEAD_DIM + kcol);
      const int vrow = c >> 2, vcol = (c & 3) * 8;   // 4 chunks per V^T row
      async_copy16(&vt[vrow][vcol],
                   Vh + (long)vrow * SEQ + kb + vcol);
    }
    wait_async0();                            // own wave's async copies done
    __syncthreads();                          // all waves' copies visible

    if (kb < kend) {
      // ---- scores: Q(16x64) x K^T from LDS ----
      v8f s0 = zero, s1 = zero;
#pragma unroll
      for (int t = 0; t < 2; ++t) {
        v16bf k0 = load_frag(&kt[0][0], HEAD_DIM, r,      t * 32 + kShift);
        v16bf k1 = load_frag(&kt[0][0], HEAD_DIM, 16 + r, t * 32 + kShift);
        s0 = wmma_bf16(qf[t], k0, s0);
        s1 = wmma_bf16(qf[t], k1, s1);
      }
      // ---- online softmax + P -> LDS (bf16, A-layout source) ----
#pragma unroll
      for (int j = 0; j < 8; ++j) {
        const int qrow = q0 + j + hi * 8;
        float v0 = (kb + r      <= qrow) ? s0[j] * scale : -1e30f;
        float v1 = (kb + 16 + r <= qrow) ? s1[j] * scale : -1e30f;
        float mx = fmaxf(v0, v1);
        mx = fmaxf(mx, __shfl_xor(mx, 1, 32));
        mx = fmaxf(mx, __shfl_xor(mx, 2, 32));
        mx = fmaxf(mx, __shfl_xor(mx, 4, 32));
        mx = fmaxf(mx, __shfl_xor(mx, 8, 32));
        const float mnew = fmaxf(mrow[j], mx);
        const float corr = __expf(mrow[j] - mnew);
        const float p0 = __expf(v0 - mnew);
        const float p1 = __expf(v1 - mnew);
        float rs = p0 + p1;
        rs += __shfl_xor(rs, 1, 32);
        rs += __shfl_xor(rs, 2, 32);
        rs += __shfl_xor(rs, 4, 32);
        rs += __shfl_xor(rs, 8, 32);
        lrow[j] = lrow[j] * corr + rs;
        mrow[j] = mnew;
#pragma unroll
        for (int nd = 0; nd < 4; ++nd) o[nd][j] *= corr;
        const int prow = j + hi * 8;
        pbuf[wave][prow][r]      = f2bf(p0);
        pbuf[wave][prow][r + 16] = f2bf(p1);
      }
      // ---- O += P(16x32) x V(32x64); same-wave LDS ops are in-order ----
      v16bf pf = load_frag(&pbuf[wave][0][0], 32, r, kShift);
#pragma unroll
      for (int nd = 0; nd < 4; ++nd) {
        v16bf vf = load_frag(&vt[0][0], 32, nd * 16 + r, kShift);
        o[nd] = wmma_bf16(pf, vf, o[nd]);
      }
    }
  }

  // ---- normalize and emit bf16 context [b, s, h*64 + d] ----
#pragma unroll
  for (int j = 0; j < 8; ++j) {
    const float inv = 1.0f / lrow[j];
    const int s = q0 + j + hi * 8;
    unsigned short* crow = ctx + ((long)b * SEQ + s) * D_MODEL + h * HEAD_DIM;
#pragma unroll
    for (int nd = 0; nd < 4; ++nd)
      crow[nd * 16 + r] = f2bf(o[nd][j] * inv);
  }
}

// ---------------- kernel 3: output projection, f32 out ----------------
__global__ void __launch_bounds__(128)
oproj_kernel(const unsigned short* __restrict__ ctx,
             const unsigned short* __restrict__ wo,
             float* __restrict__ out) {
  const int lane   = threadIdx.x & 31;
  const int wave   = threadIdx.x >> 5;
  const int r      = lane & 15;
  const int hi     = lane >> 4;
  const int kShift = hi * 8;
  const int mBase  = blockIdx.x * 16;
  const int nBase  = blockIdx.y * 256 + wave * 64;

  v8f zero = {0.f,0.f,0.f,0.f,0.f,0.f,0.f,0.f};
  v8f acc[4] = {zero, zero, zero, zero};

  for (int kk = 0; kk < D_MODEL; kk += 32) {
    __builtin_prefetch(ctx + (long)(mBase + r) * D_MODEL + kk + 64, 0, 1);
    v16bf a = load_frag(ctx, D_MODEL, mBase + r, kk + kShift);
#pragma unroll
    for (int i = 0; i < 4; ++i) {
      v16bf b = load_frag(wo, D_MODEL, nBase + i * 16 + r, kk + kShift);
      acc[i] = wmma_bf16(a, b, acc[i]);
    }
  }
#pragma unroll
  for (int i = 0; i < 4; ++i) {
#pragma unroll
    for (int j = 0; j < 8; ++j) {
      const int m = mBase + j + hi * 8;
      out[(long)m * D_MODEL + nBase + i * 16 + r] = acc[i][j];
    }
  }
}

// ---------------- launch ----------------
extern "C" void kernel_launch(void* const* d_in, const int* in_sizes, int n_in,
                              void* d_out, int out_size, void* d_ws, size_t ws_size,
                              hipStream_t stream) {
  const float* x    = (const float*)d_in[0];
  const int*   tpos = (const int*)  d_in[1];
  const float* wqkv = (const float*)d_in[2];
  const float* wo   = (const float*)d_in[3];
  float*       out  = (float*)d_out;

  // workspace carve-up (bf16 buffers), 256B aligned
  size_t off = 0;
  auto alloc = [&](size_t bytes) {
    void* p = (char*)d_ws + off;
    off += (bytes + 255) & ~(size_t)255;
    return p;
  };
  const size_t nX  = (size_t)BATCH * SEQ * D_MODEL;   // 3,145,728
  const size_t nWq = (size_t)QKV_N * D_MODEL;         // 1,769,472
  const size_t nWo = (size_t)D_MODEL * D_MODEL;       //   589,824
  const size_t nH  = (size_t)BATCH * NUM_HEADS * SEQ * HEAD_DIM;

  unsigned short* xbf   = (unsigned short*)alloc(nX  * 2);
  unsigned short* wqbf  = (unsigned short*)alloc(nWq * 2);
  unsigned short* wobf  = (unsigned short*)alloc(nWo * 2);
  unsigned short* Qb    = (unsigned short*)alloc(nH  * 2);
  unsigned short* Kb    = (unsigned short*)alloc(nH  * 2);
  unsigned short* Vt    = (unsigned short*)alloc(nH  * 2);
  unsigned short* ctx   = (unsigned short*)alloc(nX  * 2);
  (void)ws_size; (void)in_sizes; (void)n_in; (void)out_size;

  // 0) pack to bf16
  cvt_bf16_kernel<<<dim3((unsigned)((nX  / 4 + 255) / 256)), 256, 0, stream>>>(x,    xbf,  (int)(nX  / 4));
  cvt_bf16_kernel<<<dim3((unsigned)((nWq / 4 + 255) / 256)), 256, 0, stream>>>(wqkv, wqbf, (int)(nWq / 4));
  cvt_bf16_kernel<<<dim3((unsigned)((nWo / 4 + 255) / 256)), 256, 0, stream>>>(wo,   wobf, (int)(nWo / 4));

  // 1) QKV projection + RoPE + scatter (M=4096, N=2304, K=768)
  qkv_rope_kernel<<<dim3(MROWS / 16, QKV_N / 256), 128, 0, stream>>>(xbf, wqbf, tpos, Qb, Kb, Vt);

  // 2) causal flash attention (shared K/V tiles via async global->LDS)
  attn_kernel<<<dim3(SEQ / 64, BATCH * NUM_HEADS), 128, 0, stream>>>(Qb, Kb, Vt, ctx);

  // 3) output projection (M=4096, N=768, K=768), f32 out
  oproj_kernel<<<dim3(MROWS / 16, D_MODEL / 256), 128, 0, stream>>>(ctx, wobf, out);
}
